// G_Attention_27487790695180
// MI455X (gfx1250) — matrix operand order repaired and checked by reading, not compiled, not run on hardware
//
#include <hip/hip_runtime.h>
#include <hip/hip_bf16.h>

// ---------------------------------------------------------------------------
// CDNA5 (gfx1250) wave32 WMMA implementation of:
//   qkv = x @ w_qkv ; k,v = per-head LayerNorm ; scores = k^T v / n ;
//   out = (q @ scores) reshaped @ w_out + b_out
// GEMMs use v_wmma_f32_16x16x32_bf16 (fp32 acc), LDS double-buffering, and
// (when available) global_load_async_to_lds_b128 + s_wait_asynccnt.
// ---------------------------------------------------------------------------

typedef __attribute__((ext_vector_type(16))) __bf16 v16bf;
typedef __attribute__((ext_vector_type(8)))  float  v8f;
typedef int v4i __attribute__((ext_vector_type(4)));

// Per-lane WMMA operand fragment: two contiguous 16-byte runs in LDS.
union FragAB { v16bf v; uint4 q[2]; };
union FragC  { v8f   v; float f[8]; };

__device__ __forceinline__ unsigned short f32_to_bf16(float f) {
    union { float f; unsigned int u; } c; c.f = f;
    unsigned int u = c.u;
    unsigned int r = u + 0x7FFFu + ((u >> 16) & 1u);   // round-to-nearest-even
    return (unsigned short)(r >> 16);
}
__device__ __forceinline__ float bf16_to_f32(unsigned short h) {
    union { unsigned int u; float f; } c; c.u = ((unsigned int)h) << 16;
    return c.f;
}

// --- CDNA5 async global->LDS copy (ASYNCcnt-tracked), with sync fallback ---
#if defined(__has_builtin)
#if __has_builtin(__builtin_amdgcn_global_load_async_to_lds_b128)
#define HAVE_ASYNC_LDS 1
#endif
#endif

__device__ __forceinline__ void copy16_g2l(const unsigned short* g, unsigned short* l) {
#ifdef HAVE_ASYNC_LDS
    __builtin_amdgcn_global_load_async_to_lds_b128(
        (__attribute__((address_space(1))) v4i*)g,
        (__attribute__((address_space(3))) v4i*)l,
        0, 0);
#else
    *(uint4*)l = *(const uint4*)g;
#endif
}
__device__ __forceinline__ void async_wait_all() {
#ifdef HAVE_ASYNC_LDS
#if __has_builtin(__builtin_amdgcn_s_wait_asynccnt)
    __builtin_amdgcn_s_wait_asynccnt(0);
#else
    asm volatile("s_wait_asynccnt 0x0" ::: "memory");
#endif
#endif
}

// ---------------------------------------------------------------------------
// Elementwise fp32 -> bf16 conversion
// ---------------------------------------------------------------------------
__global__ void convert_f32_to_bf16(const float* __restrict__ src,
                                    unsigned short* __restrict__ dst, long n) {
    long i = (long)blockIdx.x * blockDim.x + threadIdx.x;
    long stride = (long)gridDim.x * blockDim.x;
    for (; i < n; i += stride) dst[i] = f32_to_bf16(src[i]);
}

// ---------------------------------------------------------------------------
// Generic batched bf16 GEMM: C[bz] = A[bz] @ B[bz] (row-major), fp32 acc.
// Block tile 128x64, BK=32, 256 threads = 8 wave32. Double-buffered LDS with
// async global->LDS copies for A; B is staged transposed [n][k] so the B
// fragment is two contiguous 16-byte runs per lane.
// Batch bz = (bb, hh) with independent strides so the per-(b,h) attention
// GEMM maps onto the same kernel.
// ---------------------------------------------------------------------------
#define BM 128
#define BN 64
#define BK 32
#define BKP (BK + 8)   // padded B row (80 B, keeps 16 B alignment)

template <bool OUT_BF16>
__global__ __launch_bounds__(256) void gemm_bf16_kernel(
    const unsigned short* __restrict__ A,
    const unsigned short* __restrict__ B,
    void* __restrict__ C,
    int M, int N, int K,
    int lda, int ldb, int ldc,
    int Hdiv,
    long sAb, long sAh, long sBb, long sBh, long sCb, long sCh,
    const float* __restrict__ bias, float scale) {
    __shared__ unsigned short As[2][BM][BK];    // [m][k] row-major
    __shared__ unsigned short Bs[2][BN][BKP];   // transposed [n][k]

    int bz = blockIdx.z;
    int bb = bz / Hdiv, hh = bz % Hdiv;
    const unsigned short* Ab = A + (long)bb * sAb + (long)hh * sAh;
    const unsigned short* Bb = B + (long)bb * sBb + (long)hh * sBh;
    long coff = (long)bb * sCb + (long)hh * sCh;

    int m0 = blockIdx.y * BM;
    int n0 = blockIdx.x * BN;
    int tid = threadIdx.x;
    int lane = tid & 31, wave = tid >> 5;
    int half = (lane >> 4) & 1, l15 = lane & 15;

    auto stageA = [&](int buf, int k0) {
#pragma unroll
        for (int i = 0; i < 2; ++i) {
            int f = tid * 16 + i * 8;        // 128*32 elems, 16 per thread
            int mm = f >> 5, kk = f & 31;
            copy16_g2l(Ab + (long)(m0 + mm) * lda + k0 + kk, &As[buf][mm][kk]);
        }
    };
    auto stageB = [&](int buf, int k0) {
        int f = tid * 8;                     // 32*64 elems, 8 per thread
        int kr = f >> 6, nn = f & 63;
        union { uint4 qv; unsigned short s[8]; } t;
        t.qv = *(const uint4*)(Bb + (long)(k0 + kr) * ldb + n0 + nn);
#pragma unroll
        for (int e = 0; e < 8; ++e) Bs[buf][nn + e][kr] = t.s[e];
    };

    FragC acc[4];
#pragma unroll
    for (int c = 0; c < 4; ++c)
#pragma unroll
        for (int r = 0; r < 8; ++r) acc[c].f[r] = 0.0f;

    // prologue: fill buffer 0
    stageA(0, 0);
    stageB(0, 0);
    async_wait_all();
    __syncthreads();

    int cur = 0;
    for (int k0 = 0; k0 < K; k0 += BK) {
        // overlap: start filling the other buffer while computing this one
        if (k0 + BK < K) {
            stageA(cur ^ 1, k0 + BK);
            stageB(cur ^ 1, k0 + BK);
            __builtin_prefetch(Bb + (long)(k0 + 2 * BK) * ldb + n0, 0, 1);
        }

        // A fragment: two ds_load_b128 per lane (ISA 16x32 bf16 layout)
        FragAB fa;
        const unsigned short* arow = &As[cur][wave * 16 + l15][0];
        fa.q[0] = *(const uint4*)(arow + 8 * half);
        fa.q[1] = *(const uint4*)(arow + 16 + 8 * half);

#pragma unroll
        for (int c = 0; c < 4; ++c) {
            FragAB fb;   // B fragment: two 16-byte runs at k = 8*half and 16 + 8*half
            const unsigned short* brow = &Bs[cur][c * 16 + l15][0];
            fb.q[0] = *(const uint4*)(brow + 8 * half);
            fb.q[1] = *(const uint4*)(brow + 16 + 8 * half);
            acc[c].v = __builtin_amdgcn_wmma_f32_16x16x32_bf16(
                false, fa.v, false, fb.v, (short)0, acc[c].v, false, false);
        }

        async_wait_all();   // next buffer's async copies have landed
        __syncthreads();    // everyone done reading cur before it is reused
        cur ^= 1;
    }

    // epilogue: D layout m = r + 8*half, n = lane&15
    unsigned short* C16 = (unsigned short*)C + coff;
    float* C32 = (float*)C + coff;
#pragma unroll
    for (int c = 0; c < 4; ++c) {
        int gn = n0 + c * 16 + l15;
        if (gn >= N) continue;
        float badd = bias ? bias[gn] : 0.0f;
#pragma unroll
        for (int r = 0; r < 8; ++r) {
            int gm = m0 + wave * 16 + r + 8 * half;
            if (gm >= M) continue;
            float v = acc[c].f[r] * scale + badd;
            if (OUT_BF16)
                C16[(long)gm * ldc + gn] = f32_to_bf16(v);
            else
                C32[(long)gm * ldc + gn] = v;
        }
    }
}

// ---------------------------------------------------------------------------
// Per-head LayerNorm over DH=64, in place on bf16 qkv (k and v thirds).
// One wave32 per (row, head, which); each lane owns 2 contiguous elements.
// ---------------------------------------------------------------------------
__global__ __launch_bounds__(256) void ln_kernel(
    unsigned short* __restrict__ qkv,
    const float* __restrict__ gk, const float* __restrict__ bk,
    const float* __restrict__ gv, const float* __restrict__ bv) {
    int idx = blockIdx.x * 8 + (threadIdx.x >> 5);  // row*32 + head*2 + which
    int lane = threadIdx.x & 31;
    int which = idx & 1;
    int head = (idx >> 1) & 15;
    long row = idx >> 5;

    unsigned short* p =
        qkv + row * 3072 + (long)(1 + which) * 1024 + head * 64 + lane * 2;
    unsigned int pr = *(unsigned int*)p;
    float x0 = bf16_to_f32((unsigned short)(pr & 0xFFFFu));
    float x1 = bf16_to_f32((unsigned short)(pr >> 16));

    float s = x0 + x1, s2 = x0 * x0 + x1 * x1;
#pragma unroll
    for (int o = 16; o > 0; o >>= 1) {  // wave32 butterfly reduction
        s  += __shfl_xor(s,  o, 32);
        s2 += __shfl_xor(s2, o, 32);
    }
    float mu  = s * (1.0f / 64.0f);
    float var = s2 * (1.0f / 64.0f) - mu * mu;   // biased variance
    float rstd = rsqrtf(var + 1e-5f);

    const float* g  = which ? gv : gk;
    const float* be = which ? bv : bk;
    int ci = head * 64 + lane * 2;
    float y0 = (x0 - mu) * rstd * g[ci]     + be[ci];
    float y1 = (x1 - mu) * rstd * g[ci + 1] + be[ci + 1];
    *(unsigned int*)p =
        (unsigned int)f32_to_bf16(y0) | ((unsigned int)f32_to_bf16(y1) << 16);
}

// ---------------------------------------------------------------------------
// scores[b,h] = k_ln^T @ v_ln / N  (64x64, reduce over 4096 tokens).
// One block per (b,h); stream 64-token chunks of k and v into LDS transposed
// [d][n] so both operand fragments are contiguous along the reduction axis.
// 8 waves x 2 tiles cover the 4x4 grid of 16x16 tiles.
// ---------------------------------------------------------------------------
#define SP 72   // padded row (144 B, 16 B aligned)

__global__ __launch_bounds__(256) void scores_kernel(
    const unsigned short* __restrict__ qkv,
    unsigned short* __restrict__ scores) {
    __shared__ unsigned short kT[64][SP];
    __shared__ unsigned short vT[64][SP];

    int bh = blockIdx.x;  // b*16 + h
    int b = bh >> 4, h = bh & 15;
    const unsigned short* kbase = qkv + (long)b * 4096 * 3072 + 1024 + h * 64;
    const unsigned short* vbase = qkv + (long)b * 4096 * 3072 + 2048 + h * 64;

    int tid = threadIdx.x;
    int lane = tid & 31, wave = tid >> 5;
    int half = (lane >> 4) & 1, l15 = lane & 15;

    FragC acc[2];
#pragma unroll
    for (int t = 0; t < 2; ++t)
#pragma unroll
        for (int r = 0; r < 8; ++r) acc[t].f[r] = 0.0f;

    for (int n0 = 0; n0 < 4096; n0 += 64) {
        __syncthreads();
        // stage 64x64 k and v chunks, transposed to [d][n]
#pragma unroll
        for (int i = 0; i < 2; ++i) {
            int f = tid * 16 + i * 8;
            int nn = f >> 6, dd = f & 63;
            union { uint4 qv; unsigned short s[8]; } t;
            t.qv = *(const uint4*)(kbase + (long)(n0 + nn) * 3072 + dd);
#pragma unroll
            for (int e = 0; e < 8; ++e) kT[dd + e][nn] = t.s[e];
            t.qv = *(const uint4*)(vbase + (long)(n0 + nn) * 3072 + dd);
#pragma unroll
            for (int e = 0; e < 8; ++e) vT[dd + e][nn] = t.s[e];
        }
        __syncthreads();

#pragma unroll
        for (int tt = 0; tt < 2; ++tt) {
            int t = wave * 2 + tt;
            int tr = t >> 2, tc = t & 3;
#pragma unroll
            for (int kb = 0; kb < 64; kb += 32) {
                FragAB fa, fb;
                const unsigned short* ar = &kT[tr * 16 + l15][kb];
                fa.q[0] = *(const uint4*)(ar + 8 * half);
                fa.q[1] = *(const uint4*)(ar + 16 + 8 * half);
                const unsigned short* br = &vT[tc * 16 + l15][kb];
                fb.q[0] = *(const uint4*)(br + 8 * half);
                fb.q[1] = *(const uint4*)(br + 16 + 8 * half);
                acc[tt].v = __builtin_amdgcn_wmma_f32_16x16x32_bf16(
                    false, fa.v, false, fb.v, (short)0, acc[tt].v, false, false);
            }
        }
    }

    const float inv = 1.0f / 4096.0f;
#pragma unroll
    for (int tt = 0; tt < 2; ++tt) {
        int t = wave * 2 + tt;
        int tr = t >> 2, tc = t & 3;
#pragma unroll
        for (int r = 0; r < 8; ++r) {
            int gm = tr * 16 + r + 8 * half;
            int gn = tc * 16 + l15;
            scores[(long)bh * 4096 + gm * 64 + gn] = f32_to_bf16(acc[tt].f[r] * inv);
        }
    }
}

// ---------------------------------------------------------------------------
// Host-side orchestration
// ---------------------------------------------------------------------------
extern "C" void kernel_launch(void* const* d_in, const int* in_sizes, int n_in,
                              void* d_out, int out_size, void* d_ws, size_t ws_size,
                              hipStream_t stream) {
    (void)in_sizes; (void)n_in; (void)out_size; (void)ws_size;
    const float* x     = (const float*)d_in[0];
    const float* w_qkv = (const float*)d_in[1];
    const float* gk    = (const float*)d_in[2];
    const float* bk    = (const float*)d_in[3];
    const float* gv    = (const float*)d_in[4];
    const float* bv    = (const float*)d_in[5];
    const float* w_out = (const float*)d_in[6];
    const float* b_out = (const float*)d_in[7];
    float* out = (float*)d_out;

    char* ws = (char*)d_ws;
    size_t off = 0;
    auto alloc = [&](size_t bytes) -> void* {
        void* p = ws + off;
        off += (bytes + 255) & ~(size_t)255;
        return p;
    };
    unsigned short* x_bf   = (unsigned short*)alloc(16384ull * 1024 * 2);
    unsigned short* wq_bf  = (unsigned short*)alloc(1024ull * 3072 * 2);
    unsigned short* wo_bf  = (unsigned short*)alloc(1024ull * 1024 * 2);
    unsigned short* qkv_bf = (unsigned short*)alloc(16384ull * 3072 * 2);
    unsigned short* sc_bf  = (unsigned short*)alloc(64ull * 64 * 64 * 2);
    unsigned short* at_bf  = (unsigned short*)alloc(16384ull * 1024 * 2);

    convert_f32_to_bf16<<<4096, 256, 0, stream>>>(x, x_bf, 16384ll * 1024);
    convert_f32_to_bf16<<<2048, 256, 0, stream>>>(w_qkv, wq_bf, 1024ll * 3072);
    convert_f32_to_bf16<<<1024, 256, 0, stream>>>(w_out, wo_bf, 1024ll * 1024);

    // 1) qkv = x @ w_qkv   [16384x1024]@[1024x3072] -> bf16
    dim3 g1(3072 / BN, 16384 / BM, 1);
    gemm_bf16_kernel<true><<<g1, 256, 0, stream>>>(
        x_bf, wq_bf, qkv_bf, 16384, 3072, 1024, 1024, 3072, 3072,
        1, 0, 0, 0, 0, 0, 0, nullptr, 1.0f);

    // 2) per-head LayerNorm of k and v (in place)
    ln_kernel<<<65536, 256, 0, stream>>>(qkv_bf, gk, bk, gv, bv);

    // 3) scores = k^T v / n per (b,h)
    scores_kernel<<<64, 256, 0, stream>>>(qkv_bf, sc_bf);

    // 4) attn = q @ scores per (b,h)  [4096x64]@[64x64] -> bf16 (head-major cols)
    dim3 g2(1, 4096 / BM, 64);
    gemm_bf16_kernel<true><<<g2, 256, 0, stream>>>(
        qkv_bf, sc_bf, at_bf, 4096, 64, 64, 3072, 64, 1024,
        16, 4096ll * 3072, 64, 16ll * 64 * 64, 64 * 64, 4096ll * 1024, 64,
        nullptr, 1.0f);

    // 5) out = attn @ w_out + b_out -> fp32
    dim3 g3(1024 / BN, 16384 / BM, 1);
    gemm_bf16_kernel<false><<<g3, 256, 0, stream>>>(
        at_bf, wo_bf, out, 16384, 1024, 1024, 1024, 1024, 1024,
        1, 0, 0, 0, 0, 0, 0, b_out, 1.0f);
}
